// TGNEncoder_56289841381610
// MI455X (gfx1250) — compile-verified, matrix operand order checked by reading, not run on hardware
//
#include <hip/hip_runtime.h>
#include <hip/hip_bf16.h>
#include <math.h>

// ---------------------------------------------------------------------------
// TGN encoder for MI455X (gfx1250, wave32, WMMA).
// Dense GEMMs use v_wmma_f32_16x16x32_bf16 with LDS-staged fp32->bf16 tiles:
//   - software-pipelined: next K-tile's 6x global_load_b128 issued right after
//     the barrier, latency hidden under current tile's 4 WMMAs
//   - branch-free loads (clamped addr + cndmask-zero), bf16x4 ds_store_b64
//   - distinct fragment registers -> batched ds loads, back-to-back WMMAs
//   - 128x64 block tile, wave = 16x64 (4 WMMAs / K-step sharing one A frag)
// Edge softmax-aggregation uses L2-resident atomics (max-as-uint, f32 adds).
// ---------------------------------------------------------------------------

typedef __attribute__((ext_vector_type(16))) __bf16 v16bf_t;
typedef __attribute__((ext_vector_type(4)))  __bf16 bf16x4_t;
typedef __attribute__((ext_vector_type(8)))  float  v8f_t;

#define HID 128
#define NH  4
#define CPH 32
#define G3  384
#define K0  80
#define DIN 64

// ---- helpers --------------------------------------------------------------

__device__ __forceinline__ unsigned enc_ordered(float f) {
    unsigned u = __float_as_uint(f);
    return (u & 0x80000000u) ? ~u : (u | 0x80000000u);
}
__device__ __forceinline__ float dec_ordered(unsigned u) {
    return (u & 0x80000000u) ? __uint_as_float(u & 0x7FFFFFFFu)
                             : __uint_as_float(~u);
}
__device__ __forceinline__ float sigmoidf_(float x) {
    return 1.0f / (1.0f + __expf(-x));
}
__device__ __forceinline__ float leaky02(float x) {
    return x > 0.0f ? x : 0.2f * x;
}
__device__ __forceinline__ int imin_(int a, int b) { return a < b ? a : b; }

__device__ __forceinline__ bf16x4_t pack_bf16x4(float4 v, bool ok) {
    if (!ok) { v.x = 0.f; v.y = 0.f; v.z = 0.f; v.w = 0.f; }
    bf16x4_t p;
    p[0] = (__bf16)v.x; p[1] = (__bf16)v.y;
    p[2] = (__bf16)v.z; p[3] = (__bf16)v.w;
    return p;
}

// ---- WMMA bf16 GEMM: C[M x Nout] = act(A[M x K] * W[Nout x K]^T + bias) ----
// Block: 256 threads = 8 waves. Tile: M=128 (16 rows/wave), N=64, K-step=32.
// act: 0 = none, 1 = relu.  Requires K % 4 == 0 (true here: 80, 128).

__global__ __launch_bounds__(256)
void gemm_bf16_wmma(const float* __restrict__ A, const float* __restrict__ W,
                    const float* __restrict__ bias, float* __restrict__ C,
                    int M, int K, int Nout, int act)
{
    // row stride 36 bf16 = 72 B: 8-byte aligned rows, 18-bank (odd 8B-unit) stride
    __shared__ __bf16 As[128][36];   // 9216 B
    __shared__ __bf16 Bs[64][36];    // 4608 B

    const int tid  = threadIdx.x;
    const int lane = tid & 31;
    const int wave = tid >> 5;            // 0..7
    const int m0   = blockIdx.x * 128;
    const int n0   = blockIdx.y * 64;

    // per-thread staging slots: 4 float4 of A tile, 2 float4 of B tile
    int ar[4], ak[4], br[2], bk[2];
    #pragma unroll
    for (int j = 0; j < 4; ++j) {
        int idx = tid + 256 * j; ar[j] = idx >> 3; ak[j] = (idx & 7) * 4;
    }
    #pragma unroll
    for (int j = 0; j < 2; ++j) {
        int idx = tid + 256 * j; br[j] = idx >> 3; bk[j] = (idx & 7) * 4;
    }

    // pipeline registers (live across the compute phase)
    float4 va[4], vb[2];
    bool   oka[4], okb[2];

    auto fetch = [&](int k0) {
        #pragma unroll
        for (int j = 0; j < 4; ++j) {
            int gm = m0 + ar[j], gk = k0 + ak[j];
            oka[j] = (gm < M) && (gk < K);
            va[j]  = *(const float4*)(A + (size_t)imin_(gm, M - 1) * K
                                        + imin_(gk, K - 4));
        }
        #pragma unroll
        for (int j = 0; j < 2; ++j) {
            int gn = n0 + br[j], gk = k0 + bk[j];
            okb[j] = (gn < Nout) && (gk < K);
            vb[j]  = *(const float4*)(W + (size_t)imin_(gn, Nout - 1) * K
                                        + imin_(gk, K - 4));
        }
    };

    auto stage = [&]() {
        #pragma unroll
        for (int j = 0; j < 4; ++j)
            *(bf16x4_t*)(&As[ar[j]][ak[j]]) = pack_bf16x4(va[j], oka[j]);
        #pragma unroll
        for (int j = 0; j < 2; ++j)
            *(bf16x4_t*)(&Bs[br[j]][bk[j]]) = pack_bf16x4(vb[j], okb[j]);
    };

    v8f_t acc[4] = {{}, {}, {}, {}};
    const int row = lane & 15;
    const int kb  = (lane >> 4) * 8;

    fetch(0);
    for (int k0 = 0; k0 < K; k0 += 32) {
        stage();                               // consumes pipeline regs
        __syncthreads();
        if (k0 + 32 < K) fetch(k0 + 32);       // next tile loads in flight

        // fragments per documented 16-bit wave32 layouts, distinct registers
        v16bf_t af, bf0, bf1, bf2, bf3;
        #pragma unroll
        for (int i = 0; i < 8; ++i) {
            af[i]      = As[wave * 16 + row][kb + i];
            af[i + 8]  = As[wave * 16 + row][16 + kb + i];
            bf0[i]     = Bs[row][kb + i];
            bf0[i + 8] = Bs[row][16 + kb + i];
            bf1[i]     = Bs[16 + row][kb + i];
            bf1[i + 8] = Bs[16 + row][16 + kb + i];
            bf2[i]     = Bs[32 + row][kb + i];
            bf2[i + 8] = Bs[32 + row][16 + kb + i];
            bf3[i]     = Bs[48 + row][kb + i];
            bf3[i + 8] = Bs[48 + row][16 + kb + i];
        }
        acc[0] = __builtin_amdgcn_wmma_f32_16x16x32_bf16(
            false, af, false, bf0, (short)0, acc[0], false, false);
        acc[1] = __builtin_amdgcn_wmma_f32_16x16x32_bf16(
            false, af, false, bf1, (short)0, acc[1], false, false);
        acc[2] = __builtin_amdgcn_wmma_f32_16x16x32_bf16(
            false, af, false, bf2, (short)0, acc[2], false, false);
        acc[3] = __builtin_amdgcn_wmma_f32_16x16x32_bf16(
            false, af, false, bf3, (short)0, acc[3], false, false);
        __syncthreads();
    }

    // ---- store D (16x16 f32 layout: col = lane&15, row = (lane>>4)*8 + r) ----
    const int col   = lane & 15;
    const int rbase = (lane >> 4) * 8;
    #pragma unroll
    for (int nh = 0; nh < 4; ++nh) {
        int gn = n0 + nh * 16 + col;
        if (gn >= Nout) continue;
        float bv = bias ? bias[gn] : 0.0f;
        #pragma unroll
        for (int r = 0; r < 8; ++r) {
            int gm = m0 + wave * 16 + rbase + r;
            if (gm >= M) continue;
            float v = acc[nh][r] + bv;
            if (act == 1) v = fmaxf(v, 0.0f);
            C[(size_t)gm * Nout + gn] = v;
        }
    }
}

// ---- Time2Vec (t is a single scalar; 16 values, computed once) -------------

__global__ void time2vec_kernel(const float* __restrict__ t,
                                const float* __restrict__ w0,
                                const float* __restrict__ b0,
                                const float* __restrict__ w,
                                const float* __restrict__ b,
                                float* __restrict__ t_enc)
{
    int i = threadIdx.x;
    if (i == 0)      t_enc[0] = w0[0] * t[0] + b0[0];
    else if (i < 16) t_enc[i] = sinf(w[i - 1] * t[0] + b[i - 1]);
}

// ---- concat [x | t_enc] into [N x 80] --------------------------------------

__global__ void concat_kernel(const float* __restrict__ x,
                              const float* __restrict__ t_enc,
                              float* __restrict__ xc, int total)
{
    int id = blockIdx.x * blockDim.x + threadIdx.x;
    if (id >= total) return;
    int n = id / K0, c = id % K0;
    xc[id] = (c < DIN) ? x[(size_t)n * DIN + c] : t_enc[c - DIN];
}

// ---- attention coefficients a_s, a_d : [N x H] -----------------------------

__global__ void attn_coef_kernel(const float* __restrict__ hW,
                                 const float* __restrict__ a_src,
                                 const float* __restrict__ a_dst,
                                 float* __restrict__ a_s,
                                 float* __restrict__ a_d, int total)
{
    int id = blockIdx.x * blockDim.x + threadIdx.x;
    if (id >= total) return;
    int n = id >> 2, h = id & 3;
    const float* p = hW + (size_t)n * HID + h * CPH;
    float ss = 0.0f, sd = 0.0f;
    #pragma unroll
    for (int c = 0; c < CPH; ++c) {
        float v = p[c];
        ss += v * a_src[h * CPH + c];
        sd += v * a_dst[h * CPH + c];
    }
    a_s[id] = ss;
    a_d[id] = sd;
}

// ---- edge pass 1: segment max (float encoded as ordered uint) --------------

__global__ void edge_max_kernel(const int* __restrict__ ei,
                                const float* __restrict__ a_s,
                                const float* __restrict__ a_d,
                                unsigned* __restrict__ m_u, int E_, int total)
{
    int id = blockIdx.x * blockDim.x + threadIdx.x;
    if (id >= total) return;
    int e = id >> 2, h = id & 3;
    int s = (e < E_) ? ei[e] : (e - E_);
    int d = (e < E_) ? ei[E_ + e] : (e - E_);
    float v = leaky02(a_s[s * 4 + h] + a_d[d * 4 + h]);
    atomicMax(&m_u[d * 4 + h], enc_ordered(v));
}

// ---- edge pass 2: segment sum of exp(e - m) --------------------------------

__global__ void edge_sum_kernel(const int* __restrict__ ei,
                                const float* __restrict__ a_s,
                                const float* __restrict__ a_d,
                                const unsigned* __restrict__ m_u,
                                float* __restrict__ den, int E_, int total)
{
    int id = blockIdx.x * blockDim.x + threadIdx.x;
    if (id >= total) return;
    int e = id >> 2, h = id & 3;
    int s = (e < E_) ? ei[e] : (e - E_);
    int d = (e < E_) ? ei[E_ + e] : (e - E_);
    float v = leaky02(a_s[s * 4 + h] + a_d[d * 4 + h]);
    float m = dec_ordered(m_u[d * 4 + h]);
    if (!__builtin_isfinite(m)) m = 0.0f;
    atomicAdd(&den[d * 4 + h], __expf(v - m));
}

// ---- edge pass 3: weighted feature scatter (wave per edge, 4 ch/lane) ------

__global__ __launch_bounds__(256)
void edge_scatter_kernel(const int* __restrict__ ei,
                         const float* __restrict__ a_s,
                         const float* __restrict__ a_d,
                         const unsigned* __restrict__ m_u,
                         const float* __restrict__ den,
                         const float* __restrict__ hW,
                         float* __restrict__ accb, int E_, int ET)
{
    const int lane = threadIdx.x & 31;
    const int wid  = blockIdx.x * (blockDim.x >> 5) + (threadIdx.x >> 5);
    const int nw   = gridDim.x * (blockDim.x >> 5);
    const int h    = lane >> 3;           // head 0..3
    const int cg   = (lane & 7) * 4;      // channel group within head

    for (int e = wid; e < ET; e += nw) {
        int s = (e < E_) ? ei[e] : (e - E_);
        int d = (e < E_) ? ei[E_ + e] : (e - E_);
        float v = leaky02(a_s[s * 4 + h] + a_d[d * 4 + h]);
        float m = dec_ordered(m_u[d * 4 + h]);
        if (!__builtin_isfinite(m)) m = 0.0f;
        float w = __expf(v - m) / den[d * 4 + h];
        const float4 hv = *(const float4*)(hW + (size_t)s * HID + h * CPH + cg);
        float* ap = accb + (size_t)d * HID + h * CPH + cg;
        atomicAdd(ap + 0, hv.x * w);
        atomicAdd(ap + 1, hv.y * w);
        atomicAdd(ap + 2, hv.z * w);
        atomicAdd(ap + 3, hv.w * w);
    }
}

// ---- bias + ELU ------------------------------------------------------------

__global__ void bias_elu_kernel(const float* __restrict__ accb,
                                const float* __restrict__ bias,
                                float* __restrict__ out, int total)
{
    int id = blockIdx.x * blockDim.x + threadIdx.x;
    if (id >= total) return;
    float v = accb[id] + bias[id & (HID - 1)];
    out[id] = v > 0.0f ? v : (__expf(v) - 1.0f);
}

// ---- LayerNorm, wave per node (128 = 4 values/lane, wave32 shfl reduce) ----

__global__ __launch_bounds__(256)
void layernorm_kernel(float* __restrict__ hbuf,
                      const float* __restrict__ g,
                      const float* __restrict__ b, int Nn)
{
    const int lane = threadIdx.x & 31;
    const int wid  = blockIdx.x * (blockDim.x >> 5) + (threadIdx.x >> 5);
    const int nw   = gridDim.x * (blockDim.x >> 5);

    for (int n = wid; n < Nn; n += nw) {
        float* p = hbuf + (size_t)n * HID;
        float4 v = *(const float4*)(p + lane * 4);
        float s  = v.x + v.y + v.z + v.w;
        float sq = v.x * v.x + v.y * v.y + v.z * v.z + v.w * v.w;
        #pragma unroll
        for (int o = 16; o > 0; o >>= 1) {
            s  += __shfl_xor(s, o);
            sq += __shfl_xor(sq, o);
        }
        float mu  = s * (1.0f / 128.0f);
        float var = sq * (1.0f / 128.0f) - mu * mu;
        float inv = rsqrtf(var + 1e-5f);
        int c = lane * 4;
        float4 o4;
        o4.x = (v.x - mu) * inv * g[c + 0] + b[c + 0];
        o4.y = (v.y - mu) * inv * g[c + 1] + b[c + 1];
        o4.z = (v.z - mu) * inv * g[c + 2] + b[c + 2];
        o4.w = (v.w - mu) * inv * g[c + 3] + b[c + 3];
        *(float4*)(p + lane * 4) = o4;
    }
}

// ---- GRU gates -------------------------------------------------------------

__global__ void gru_gate_kernel(const float* __restrict__ gi,
                                const float* __restrict__ gh,
                                const float* __restrict__ mem,
                                float* __restrict__ out, int total)
{
    int id = blockIdx.x * blockDim.x + threadIdx.x;
    if (id >= total) return;
    int n = id >> 7, c = id & 127;
    const float* pi = gi + (size_t)n * G3;
    const float* ph = gh + (size_t)n * G3;
    float r  = sigmoidf_(pi[c]       + ph[c]);
    float z  = sigmoidf_(pi[128 + c] + ph[128 + c]);
    float nn = tanhf(pi[256 + c] + r * ph[256 + c]);
    out[id] = (1.0f - z) * nn + z * mem[id];
}

// ---------------------------------------------------------------------------

static inline int cdiv(int a, int b) { return (a + b - 1) / b; }

extern "C" void kernel_launch(void* const* d_in, const int* in_sizes, int n_in,
                              void* d_out, int out_size, void* d_ws, size_t ws_size,
                              hipStream_t stream)
{
    const float* x      = (const float*)d_in[0];
    const float* t      = (const float*)d_in[1];
    const float* tv_w0  = (const float*)d_in[2];
    const float* tv_b0  = (const float*)d_in[3];
    const float* tv_w   = (const float*)d_in[4];
    const float* tv_b   = (const float*)d_in[5];
    const float* proj_W = (const float*)d_in[6];
    const float* proj_b = (const float*)d_in[7];
    const float* gW[2]  = {(const float*)d_in[8],  (const float*)d_in[12]};
    const float* gAs[2] = {(const float*)d_in[9],  (const float*)d_in[13]};
    const float* gAd[2] = {(const float*)d_in[10], (const float*)d_in[14]};
    const float* gB[2]  = {(const float*)d_in[11], (const float*)d_in[15]};
    const float* ln_g   = (const float*)d_in[16];
    const float* ln_b   = (const float*)d_in[17];
    const float* gWih   = (const float*)d_in[18];
    const float* gWhh   = (const float*)d_in[19];
    const float* gbih   = (const float*)d_in[20];
    const float* gbhh   = (const float*)d_in[21];
    const float* mem    = (const float*)d_in[22];
    const int*   ei     = (const int*)d_in[23];

    const int Nn = in_sizes[0] / DIN;      // 50000
    const int Er = in_sizes[23] / 2;       // 800000
    const int ET = Er + Nn;                // edges + self loops

    // ---- carve workspace ----
    char* ws = (char*)d_ws;
    size_t off = 0;
    auto carve = [&](size_t bytes) -> char* {
        char* p = ws + off;
        off += (bytes + 255) & ~(size_t)255;
        return p;
    };
    float*    t_enc = (float*)   carve(16 * sizeof(float));
    float*    xc    = (float*)   carve((size_t)Nn * K0  * sizeof(float));
    float*    hbuf  = (float*)   carve((size_t)Nn * HID * sizeof(float));
    float*    hW    = (float*)   carve((size_t)Nn * HID * sizeof(float));
    float*    accb  = (float*)   carve((size_t)Nn * HID * sizeof(float));
    float*    a_s   = (float*)   carve((size_t)Nn * NH  * sizeof(float));
    float*    a_d   = (float*)   carve((size_t)Nn * NH  * sizeof(float));
    unsigned* m_u   = (unsigned*)carve((size_t)Nn * NH  * sizeof(unsigned));
    float*    den   = (float*)   carve((size_t)Nn * NH  * sizeof(float));
    float*    gi    = (float*)   carve((size_t)Nn * G3  * sizeof(float));
    float*    gh    = (float*)   carve((size_t)Nn * G3  * sizeof(float));
    (void)ws_size; (void)n_in; (void)out_size;

    // ---- Time2Vec + concat + input projection (relu) ----
    time2vec_kernel<<<1, 32, 0, stream>>>(t, tv_w0, tv_b0, tv_w, tv_b, t_enc);
    concat_kernel<<<cdiv(Nn * K0, 256), 256, 0, stream>>>(x, t_enc, xc, Nn * K0);
    gemm_bf16_wmma<<<dim3(cdiv(Nn, 128), cdiv(HID, 64)), 256, 0, stream>>>(
        xc, proj_W, proj_b, hbuf, Nn, K0, HID, /*act=*/1);

    // ---- two GAT layers ----
    for (int l = 0; l < 2; ++l) {
        gemm_bf16_wmma<<<dim3(cdiv(Nn, 128), cdiv(HID, 64)), 256, 0, stream>>>(
            hbuf, gW[l], nullptr, hW, Nn, HID, HID, /*act=*/0);
        attn_coef_kernel<<<cdiv(Nn * NH, 256), 256, 0, stream>>>(
            hW, gAs[l], gAd[l], a_s, a_d, Nn * NH);
        hipMemsetAsync(m_u,  0, (size_t)Nn * NH  * sizeof(unsigned), stream);
        hipMemsetAsync(den,  0, (size_t)Nn * NH  * sizeof(float),    stream);
        hipMemsetAsync(accb, 0, (size_t)Nn * HID * sizeof(float),    stream);
        edge_max_kernel<<<cdiv(ET * NH, 256), 256, 0, stream>>>(
            ei, a_s, a_d, m_u, Er, ET * NH);
        edge_sum_kernel<<<cdiv(ET * NH, 256), 256, 0, stream>>>(
            ei, a_s, a_d, m_u, den, Er, ET * NH);
        edge_scatter_kernel<<<cdiv(ET, 8), 256, 0, stream>>>(
            ei, a_s, a_d, m_u, den, hW, accb, Er, ET);
        bias_elu_kernel<<<cdiv(Nn * HID, 256), 256, 0, stream>>>(
            accb, gB[l], hbuf, Nn * HID);
    }

    // ---- LayerNorm (in place on hbuf) ----
    layernorm_kernel<<<cdiv(Nn, 8), 256, 0, stream>>>(hbuf, ln_g, ln_b, Nn);

    // ---- GRU cell: gi = h @ Wih^T + bih ; gh = memory @ Whh^T + bhh ----
    gemm_bf16_wmma<<<dim3(cdiv(Nn, 128), cdiv(G3, 64)), 256, 0, stream>>>(
        hbuf, gWih, gbih, gi, Nn, HID, G3, /*act=*/0);
    gemm_bf16_wmma<<<dim3(cdiv(Nn, 128), cdiv(G3, 64)), 256, 0, stream>>>(
        mem, gWhh, gbhh, gh, Nn, HID, G3, /*act=*/0);
    gru_gate_kernel<<<cdiv(Nn * HID, 256), 256, 0, stream>>>(
        gi, gh, mem, (float*)d_out, Nn * HID);
}
